// get_model_12489764896768
// MI455X (gfx1250) — matrix operand order repaired and checked by reading, not compiled, not run on hardware
//
#include <hip/hip_runtime.h>
#include <vector>
#include <initializer_list>
#include <cstddef>
#include <cstdint>

typedef _Float16 h16;
typedef __attribute__((ext_vector_type(8)))  _Float16 v8h;
typedef __attribute__((ext_vector_type(16))) _Float16 v16h;
typedef __attribute__((ext_vector_type(8)))  float    v8f;

#define BN_SCALE 0.99999500003749978f

static inline int pad32(int c){ return (c + 31) & ~31; }
static inline unsigned cdivz(size_t a, unsigned b){ return (unsigned)((a + b - 1) / b); }

// ---------------------------------------------------------------------------
// Elementwise / setup kernels
// ---------------------------------------------------------------------------

// Convert f32 weights (O x C) to f16 (Opad x ldw), zero-padding both the
// column tail (c >= C) and the row tail (o >= O) so the GEMM can load A/B
// tiles unconditionally.
__global__ void k_w_to_h(const float* __restrict__ W, h16* __restrict__ Wh,
                         int O, int C, int ldw, int Opad){
  size_t i = blockIdx.x * (size_t)blockDim.x + threadIdx.x;
  size_t tot = (size_t)Opad * ldw;
  if (i >= tot) return;
  int c = (int)(i % ldw); int o = (int)(i / ldw);
  Wh[i] = (c < C && o < O) ? (h16)W[(size_t)o * C + c] : (h16)0.f;
}

// pointcloud (B,3,N) f32 -> xyz (B,N,3) f32
__global__ void k_xyz_cf(const float* __restrict__ pc, float* __restrict__ xyz,
                         int Nt, size_t total){
  size_t i = blockIdx.x * (size_t)blockDim.x + threadIdx.x;
  if (i >= total) return;
  int c = (int)(i % 3); size_t r = i / 3;
  int n = (int)(r % Nt); int b = (int)(r / Nt);
  xyz[i] = pc[((size_t)b * 3 + c) * Nt + n];
}

// pointcloud (B,C,N) f32 -> features rows (B*N, ld) f16, zero pad
__global__ void k_feat_cf(const float* __restrict__ pc, h16* __restrict__ f,
                          int Nt, int C, int ld, size_t total){
  size_t i = blockIdx.x * (size_t)blockDim.x + threadIdx.x;
  if (i >= total) return;
  size_t row = i / ld; int c = (int)(i % ld);
  int b = (int)(row / Nt); int n = (int)(row % Nt);
  f[i] = (c < C) ? (h16)pc[((size_t)b * C + c) * Nt + n] : (h16)0.f;
}

// write xyz (f32, rows*3) into first 3 f16 columns of dst
__global__ void k_cat_xyz(const float* __restrict__ xyz, h16* __restrict__ dst,
                          int ldo, size_t total /* = rows*3 */){
  size_t i = blockIdx.x * (size_t)blockDim.x + threadIdx.x;
  if (i >= total) return;
  size_t row = i / 3; int c = (int)(i % 3);
  dst[row * (size_t)ldo + c] = (h16)xyz[i];
}

__global__ void k_copy_cols(const h16* __restrict__ src, int ldi, int Cc,
                            h16* __restrict__ dst, int ldo, int off, size_t total){
  size_t i = blockIdx.x * (size_t)blockDim.x + threadIdx.x;
  if (i >= total) return;
  size_t row = i / Cc; int c = (int)(i % Cc);
  dst[row * (size_t)ldo + off + c] = src[row * (size_t)ldi + c];
}

// broadcast single row per batch (S==1 interpolation path)
__global__ void k_bcast_cols(const h16* __restrict__ p2, int ld2, int C2,
                             h16* __restrict__ dst, int ldo, int off, int N1,
                             size_t total){
  size_t i = blockIdx.x * (size_t)blockDim.x + threadIdx.x;
  if (i >= total) return;
  size_t row = i / C2; int c = (int)(i % C2); int b = (int)(row / N1);
  dst[row * (size_t)ldo + off + c] = p2[(size_t)b * ld2 + c];
}

// ---------------------------------------------------------------------------
// Point sampling / grouping kernels
// ---------------------------------------------------------------------------

__global__ __launch_bounds__(256) void k_fps(const float* __restrict__ xyz,
                                             int Np, int S, int* __restrict__ fidx){
  __shared__ float dist[2048];
  __shared__ float rv[256];
  __shared__ int   ri[256];
  __shared__ int   far_s;
  int b = blockIdx.x;
  const float* xb = xyz + (size_t)b * Np * 3;
  int t = threadIdx.x;
  for (int i = t; i < Np; i += 256) dist[i] = 1e10f;
  if (t == 0) far_s = 0;
  __syncthreads();
  for (int it = 0; it < S; ++it){
    int f = far_s;
    if (t == 0) fidx[b * S + it] = f;
    float cx = xb[f*3+0], cy = xb[f*3+1], cz = xb[f*3+2];
    float bv = -1.f; int bi = 0;
    for (int i = t; i < Np; i += 256){
      float dx = xb[i*3+0]-cx, dy = xb[i*3+1]-cy, dz = xb[i*3+2]-cz;
      float d = dx*dx + dy*dy + dz*dz;
      float nd = fminf(dist[i], d); dist[i] = nd;
      if (nd > bv) { bv = nd; bi = i; }
    }
    rv[t] = bv; ri[t] = bi;
    __syncthreads();
    for (int s = 128; s > 0; s >>= 1){
      if (t < s){
        if (rv[t+s] > rv[t] || (rv[t+s] == rv[t] && ri[t+s] < ri[t])){
          rv[t] = rv[t+s]; ri[t] = ri[t+s];
        }
      }
      __syncthreads();
    }
    if (t == 0) far_s = ri[0];
    __syncthreads();
  }
}

__global__ void k_gather_xyz(const float* __restrict__ xyz, const int* __restrict__ fidx,
                             int Np, int S, float* __restrict__ nxyz, size_t total){
  size_t i = blockIdx.x * (size_t)blockDim.x + threadIdx.x;
  if (i >= total) return;
  int b = (int)(i / S);
  int p = fidx[i];
  for (int c = 0; c < 3; ++c)
    nxyz[i*3 + c] = xyz[((size_t)b * Np + p) * 3 + c];
}

__global__ void k_ball_query(const float* __restrict__ xyz, const float* __restrict__ nxyz,
                             int Np, int S, int K, float r2, int* __restrict__ idx,
                             size_t total){
  size_t i = blockIdx.x * (size_t)blockDim.x + threadIdx.x;
  if (i >= total) return;
  int b = (int)(i / S);
  const float* xb = xyz + (size_t)b * Np * 3;
  float cx = nxyz[i*3], cy = nxyz[i*3+1], cz = nxyz[i*3+2];
  int* out = idx + i * (size_t)K;
  int cnt = 0;
  for (int p = 0; p < Np && cnt < K; ++p){
    float dx = xb[p*3]-cx, dy = xb[p*3+1]-cy, dz = xb[p*3+2]-cz;
    if (dx*dx + dy*dy + dz*dz <= r2) out[cnt++] = p;
  }
  int f0 = (cnt > 0) ? out[0] : 0;
  for (; cnt < K; ++cnt) out[cnt] = f0;
}

// g[b,s,k,:] = concat(xyz[p]-new_xyz[s], pts[p]); zero pad columns
__global__ void k_group(const float* __restrict__ xyz, const float* __restrict__ nxyz,
                        const h16* __restrict__ pts, int ldp, const int* __restrict__ idx,
                        int Np, int S, int K, int Cp, int ldg,
                        h16* __restrict__ g, size_t total){
  size_t i = blockIdx.x * (size_t)blockDim.x + threadIdx.x;
  if (i >= total) return;
  size_t row = i / ldg; int c = (int)(i % ldg);
  size_t bs = row / K; int b = (int)(bs / S);
  int p = idx[row];
  h16 v = (h16)0.f;
  if (c < 3)            v = (h16)(xyz[((size_t)b*Np + p)*3 + c] - nxyz[bs*3 + c]);
  else if (c < 3 + Cp)  v = pts[((size_t)b*Np + p) * ldp + (c - 3)];
  g[i] = v;
}

// group_all: g[b,n,:] = concat(xyz[b,n], pts[b,n])
__global__ void k_group_all(const float* __restrict__ xyz, const h16* __restrict__ pts,
                            int ldp, int Np, int Cp, int ldg,
                            h16* __restrict__ g, size_t total){
  size_t i = blockIdx.x * (size_t)blockDim.x + threadIdx.x;
  if (i >= total) return;
  size_t row = i / ldg; int c = (int)(i % ldg);
  h16 v = (h16)0.f;
  if (c < 3)           v = (h16)xyz[row*3 + c];
  else if (c < 3 + Cp) v = pts[row * (size_t)ldp + (c - 3)];
  g[i] = v;
}

__global__ void k_maxpool(const h16* __restrict__ in, int ld, int K,
                          h16* __restrict__ out, size_t total){
  size_t i = blockIdx.x * (size_t)blockDim.x + threadIdx.x;
  if (i >= total) return;
  size_t ro = i / ld; int c = (int)(i % ld);
  const h16* p = in + (ro * (size_t)K) * ld + c;
  float m = -3e38f;
  for (int k = 0; k < K; ++k) m = fmaxf(m, (float)p[(size_t)k * ld]);
  out[i] = (h16)m;
}

// ---------------------------------------------------------------------------
// 3-NN inverse-distance interpolation
// ---------------------------------------------------------------------------

__global__ void k_knn3(const float* __restrict__ x1, const float* __restrict__ x2,
                       int N1, int S, int* __restrict__ ki, float* __restrict__ kw,
                       size_t total){
  size_t i = blockIdx.x * (size_t)blockDim.x + threadIdx.x;
  if (i >= total) return;
  int b = (int)(i / N1);
  const float* a  = x1 + i * 3;
  const float* xb = x2 + (size_t)b * S * 3;
  float ax = a[0], ay = a[1], az = a[2];
  float d0 = 3e38f, d1 = 3e38f, d2 = 3e38f; int i0 = 0, i1 = 0, i2 = 0;
  for (int j = 0; j < S; ++j){
    float dx = xb[j*3]-ax, dy = xb[j*3+1]-ay, dz = xb[j*3+2]-az;
    float d = dx*dx + dy*dy + dz*dz;
    if (d < d0)      { d2=d1; i2=i1; d1=d0; i1=i0; d0=d; i0=j; }
    else if (d < d1) { d2=d1; i2=i1; d1=d; i1=j; }
    else if (d < d2) { d2=d; i2=j; }
  }
  float w0 = 1.f/(d0 + 1e-8f), w1 = 1.f/(d1 + 1e-8f), w2 = 1.f/(d2 + 1e-8f);
  float s = w0 + w1 + w2;
  ki[i*3]=i0; ki[i*3+1]=i1; ki[i*3+2]=i2;
  kw[i*3]=w0/s; kw[i*3+1]=w1/s; kw[i*3+2]=w2/s;
}

__global__ void k_interp(const h16* __restrict__ p2, int ld2, int C2,
                         const int* __restrict__ ki, const float* __restrict__ kw,
                         int S, int N1, h16* __restrict__ dst, int ldo, int off,
                         size_t total){
  size_t i = blockIdx.x * (size_t)blockDim.x + threadIdx.x;
  if (i >= total) return;
  size_t row = i / C2; int c = (int)(i % C2); int b = (int)(row / N1);
  size_t k3 = row * 3;
  size_t base = (size_t)b * S;
  float v = kw[k3+0] * (float)p2[(base + ki[k3+0]) * (size_t)ld2 + c]
          + kw[k3+1] * (float)p2[(base + ki[k3+1]) * (size_t)ld2 + c]
          + kw[k3+2] * (float)p2[(base + ki[k3+2]) * (size_t)ld2 + c];
  dst[row * (size_t)ldo + off + c] = (h16)v;
}

// ---------------------------------------------------------------------------
// WMMA GEMM:  Y[M,O] = act( (X[M,Cs] * W[O,Cs]^T + bias) * scale )
//   One 16x32 output tile per wave: one A load, two B loads, two
//   v_wmma_f32_16x16x32_f16 per k-step.  W is padded with zero rows/cols so
//   the inner loop has NO divergence (EXEC stays all-1s for every WMMA).
// ---------------------------------------------------------------------------

__global__ __launch_bounds__(256) void k_gemm(
    const h16* __restrict__ X, int ldx,
    const h16* __restrict__ W, int ldw,
    const float* __restrict__ bias,
    void* __restrict__ Y, int ldy,
    int M, int O, float scale, int relu, int outf32)
{
  const int Cs = ldw;
  int tilesM = M >> 4;
  int tilesO = (O + 31) >> 5;
  int tile = blockIdx.x * (blockDim.x >> 5) + ((int)threadIdx.x >> 5);
  if (tile >= tilesM * tilesO) return;          // wave-uniform early exit
  int mt = tile % tilesM;
  int ot = tile / tilesM;
  int m0 = mt << 4, o0 = ot << 5;
  int lane = threadIdx.x & 31;
  int ln = lane & 15;
  int hi = lane >> 4;

  // A: 16x32 f16, lanes 0-15 hold K 0-15, lanes 16-31 hold K 16-31
  const h16* arow  = X + (size_t)(m0 + ln) * ldx + (hi ? 16 : 0);
  // B: 32x16 f16 (=W^T tile), lane n holds K 0-15 (lanes 0-15) or 16-31 (16-31)
  const h16* brow0 = W + (size_t)(o0 + ln) * ldw + (hi ? 16 : 0);
  const h16* brow1 = brow0 + (size_t)16 * ldw;

  v8f acc0 = {}, acc1 = {};
  for (int k0 = 0; k0 < Cs; k0 += 32){
    v8h alo = *(const v8h*)(arow + k0);
    v8h ahi = *(const v8h*)(arow + k0 + 8);
    v16h a;
#pragma unroll
    for (int i = 0; i < 8; ++i){ a[i] = alo[i]; a[i + 8] = ahi[i]; }
    v16h b0 = *(const v16h*)(brow0 + k0);
    v16h b1 = *(const v16h*)(brow1 + k0);
    acc0 = __builtin_amdgcn_wmma_f32_16x16x32_f16(false, a, false, b0,
                                                  (short)0, acc0, false, false);
    acc1 = __builtin_amdgcn_wmma_f32_16x16x32_f16(false, a, false, b1,
                                                  (short)0, acc1, false, false);
  }
  int oc0 = o0 + ln;
  int oc1 = oc0 + 16;
  if (oc0 < O){
    float bb = bias[oc0];
#pragma unroll
    for (int r = 0; r < 8; ++r){
      int row = m0 + r + (hi << 3);            // C/D layout: VGPR r -> M=r / M=8+r
      float v = (acc0[r] + bb) * scale;
      if (relu) v = fmaxf(v, 0.f);
      if (outf32) ((float*)Y)[(size_t)row * ldy + oc0] = v;
      else        ((h16*) Y)[(size_t)row * ldy + oc0] = (h16)v;
    }
  }
  if (oc1 < O){
    float bb = bias[oc1];
#pragma unroll
    for (int r = 0; r < 8; ++r){
      int row = m0 + r + (hi << 3);
      float v = (acc1[r] + bb) * scale;
      if (relu) v = fmaxf(v, 0.f);
      if (outf32) ((float*)Y)[(size_t)row * ldy + oc1] = v;
      else        ((h16*) Y)[(size_t)row * ldy + oc1] = (h16)v;
    }
  }
}

// ---------------------------------------------------------------------------
// Host orchestration
// ---------------------------------------------------------------------------

struct LayerDesc { int O, C, ldw, Opad; const float* W; const float* b; h16* Wh; };

static void run_gemm(hipStream_t st, const h16* X, int ldx, const LayerDesc& L,
                     void* Y, int ldy, int M, float scale, bool relu, bool outf32){
  int tiles  = (M >> 4) * ((L.O + 31) >> 5);
  int blocks = (tiles + 7) / 8;
  k_gemm<<<blocks, 256, 0, st>>>(X, ldx, L.Wh, L.ldw, L.b, Y, ldy, M, L.O,
                                 scale, relu ? 1 : 0, outf32 ? 1 : 0);
}

struct Ctx {
  hipStream_t st;
  h16 *A, *B2;          // ping-pong slabs
  int *fpsb, *bqb, *ki;
  float *kw;
};

static const int BT = 4;

static void sa_stage(const Ctx& C, const float* xyz, int Np, const h16* pts, int ldp, int Cp,
                     int S, int K, float radius, const LayerDesc* L, int nL,
                     float* nxyz, h16* out){
  k_fps<<<BT, 256, 0, C.st>>>(xyz, Np, S, C.fpsb);
  size_t t1 = (size_t)BT * S;
  k_gather_xyz<<<cdivz(t1,256),256,0,C.st>>>(xyz, C.fpsb, Np, S, nxyz, t1);
  k_ball_query<<<cdivz(t1,256),256,0,C.st>>>(xyz, nxyz, Np, S, K, radius*radius, C.bqb, t1);
  int ldg = pad32(3 + Cp);
  size_t R = (size_t)BT * S * K;
  size_t tg = R * ldg;
  k_group<<<cdivz(tg,256),256,0,C.st>>>(xyz, nxyz, pts, ldp, C.bqb, Np, S, K, Cp, ldg, C.A, tg);
  const h16* cur = C.A; int ld = ldg;
  for (int i = 0; i < nL; ++i){
    h16* dst = (i & 1) ? C.A : C.B2;
    run_gemm(C.st, cur, ld, L[i], dst, pad32(L[i].O), (int)R, BN_SCALE, true, false);
    cur = dst; ld = pad32(L[i].O);
  }
  size_t tp = (size_t)BT * S * ld;
  k_maxpool<<<cdivz(tp,256),256,0,C.st>>>(cur, ld, K, out, tp);
}

static void sa_groupall(const Ctx& C, const float* xyz, int Np, const h16* pts, int ldp, int Cp,
                        const LayerDesc* L, int nL, h16* out){
  int ldg = pad32(3 + Cp);
  size_t R = (size_t)BT * Np;
  size_t tg = R * ldg;
  k_group_all<<<cdivz(tg,256),256,0,C.st>>>(xyz, pts, ldp, Np, Cp, ldg, C.A, tg);
  const h16* cur = C.A; int ld = ldg;
  for (int i = 0; i < nL; ++i){
    h16* dst = (i & 1) ? C.A : C.B2;
    run_gemm(C.st, cur, ld, L[i], dst, pad32(L[i].O), (int)R, BN_SCALE, true, false);
    cur = dst; ld = pad32(L[i].O);
  }
  size_t tp = (size_t)BT * ld;   // rows_out = BT (S==1), pool over Np
  k_maxpool<<<cdivz(tp,256),256,0,C.st>>>(cur, ld, Np, out, tp);
}

static void fp_stage(const Ctx& C, const float* xyz1, int N1, const float* xyz2, int S2,
                     const h16* p1, int ld1, int C1, const h16* p2, int ld2, int C2,
                     const LayerDesc* L, int nL, h16* out){
  size_t R = (size_t)BT * N1;
  int ldc = pad32(C1 + C2);
  k_copy_cols<<<cdivz(R*(size_t)C1,256),256,0,C.st>>>(p1, ld1, C1, C.A, ldc, 0, R*(size_t)C1);
  if (S2 == 1){
    k_bcast_cols<<<cdivz(R*(size_t)C2,256),256,0,C.st>>>(p2, ld2, C2, C.A, ldc, C1, N1, R*(size_t)C2);
  } else {
    k_knn3<<<cdivz(R,256),256,0,C.st>>>(xyz1, xyz2, N1, S2, C.ki, C.kw, R);
    k_interp<<<cdivz(R*(size_t)C2,256),256,0,C.st>>>(p2, ld2, C2, C.ki, C.kw, S2, N1,
                                                     C.A, ldc, C1, R*(size_t)C2);
  }
  const h16* cur = C.A; int ld = ldc;
  for (int i = 0; i < nL; ++i){
    h16* dst = (i == nL - 1) ? out : ((i & 1) ? C.A : C.B2);
    run_gemm(C.st, cur, ld, L[i], dst, pad32(L[i].O), (int)R, BN_SCALE, true, false);
    cur = dst; ld = pad32(L[i].O);
  }
}

extern "C" void kernel_launch(void* const* d_in, const int* in_sizes, int n_in,
                              void* d_out, int out_size, void* d_ws, size_t ws_size,
                              hipStream_t stream){
  (void)in_sizes; (void)n_in; (void)out_size; (void)ws_size;
  const int Nt = 2048;
  char* base = (char*)d_ws; size_t cur = 0;
  auto alloc = [&](size_t bytes)->void*{
    size_t a = (cur + 255) & ~(size_t)255; cur = a + bytes; return (void*)(base + a);
  };

  const float* pc = (const float*)d_in[0];
  int ip = 1;
  std::vector<LayerDesc> Ls; Ls.reserve(160);
  auto add_mlp = [&](int cin, std::initializer_list<int> dims)->int{
    int st = (int)Ls.size();
    for (int co : dims){
      LayerDesc d; d.C = cin; d.O = co; d.ldw = pad32(cin);
      d.Opad = (co + 31) & ~31;
      d.W = (const float*)d_in[ip++]; d.b = (const float*)d_in[ip++];
      d.Wh = (h16*)alloc((size_t)d.Opad * d.ldw * sizeof(h16));
      Ls.push_back(d); cin = co;
    }
    return st;
  };

  // Parameter groups in setup_inputs insertion order
  int sa01 = add_mlp(6,    {64,64,128});
  int sa02 = add_mlp(131,  {128,128,256});
  int sa03 = add_mlp(259,  {256,512,1024});
  int fp03 = add_mlp(1280, {256,256});
  int fp02 = add_mlp(384,  {256,128});
  int fp01 = add_mlp(134,  {128,128,128});
  int sa04 = add_mlp(6,    {64,64,128});
  int sa05 = add_mlp(131,  {128,128,256});
  int sa06 = add_mlp(259,  {256,512,1024});
  int fp04 = add_mlp(1280, {256,256});
  int fp05 = add_mlp(384,  {256,128});
  int fp06 = add_mlp(134,  {128,128,128});
  int conv00 = add_mlp(128, {128});
  int conv01 = add_mlp(128, {128});
  int bsa1[4],bsa2[4],bsa3[4],bfp3[4],bfp2[4],bfp1[4],bc0[4],bc1[4],bc2[4],bc3[4],bc4[4];
  for (int b = 0; b < 4; ++b){
    bsa1[b] = add_mlp(259,  {64,64,128});
    bsa2[b] = add_mlp(131,  {128,128,256});
    bsa3[b] = add_mlp(259,  {256,512,1024});
    bfp3[b] = add_mlp(1280, {256,256});
    bfp2[b] = add_mlp(384,  {256,128});
    bfp1[b] = add_mlp(387,  {128,128,128});
    bc0[b]  = add_mlp(128,  {512});
    bc1[b]  = add_mlp(515,  {128});
    bc2[b]  = add_mlp(131,  {32});
    bc3[b]  = add_mlp(35,   {1});
    bc4[b]  = add_mlp(1,    {1});
  }

  // Convert / zero-pad all weights to f16 once
  for (auto& L : Ls){
    size_t tot = (size_t)L.Opad * L.ldw;
    k_w_to_h<<<cdivz(tot,256),256,0,stream>>>(L.W, L.Wh, L.O, L.C, L.ldw, L.Opad);
  }

  // Persistent buffers
  float* xyz0  = (float*)alloc((size_t)BT*Nt*3*4);
  h16*   feat0 = (h16*)  alloc((size_t)BT*Nt*32*2);
  h16*   cat6  = (h16*)  alloc((size_t)BT*Nt*32*2);
  float* l1x = (float*)alloc((size_t)BT*512*3*4);
  h16*   l1p = (h16*)  alloc((size_t)BT*512*128*2);
  float* l2x = (float*)alloc((size_t)BT*128*3*4);
  h16*   l2p = (h16*)  alloc((size_t)BT*128*256*2);
  h16*   l3p = (h16*)  alloc((size_t)BT*1024*2);
  h16*   l2pf= (h16*)  alloc((size_t)BT*128*256*2);   // l2_points0 (persists into branches)
  h16*   l1pf= (h16*)  alloc((size_t)BT*512*128*2);
  h16*   l0p0= (h16*)  alloc((size_t)BT*Nt*128*2);
  float* m1x = (float*)alloc((size_t)BT*512*3*4);
  h16*   m1p = (h16*)  alloc((size_t)BT*512*128*2);
  float* m2x = (float*)alloc((size_t)BT*128*3*4);
  h16*   m2p = (h16*)  alloc((size_t)BT*128*256*2);
  h16*   m3p = (h16*)  alloc((size_t)BT*1024*2);
  h16*   m2pf= (h16*)  alloc((size_t)BT*128*256*2);
  h16*   m1pf= (h16*)  alloc((size_t)BT*512*128*2);
  h16*   l0p1= (h16*)  alloc((size_t)BT*Nt*128*2);
  h16*   net = (h16*)  alloc((size_t)BT*Nt*256*2);
  h16*  cat259=(h16*)  alloc((size_t)BT*Nt*288*2);
  // branch-reusable
  float* g1x = (float*)alloc((size_t)BT*512*3*4);
  h16*   g1p = (h16*)  alloc((size_t)BT*512*128*2);
  float* g2x = (float*)alloc((size_t)BT*128*3*4);
  h16*   g2p = (h16*)  alloc((size_t)BT*128*256*2);
  h16*   g3p = (h16*)  alloc((size_t)BT*1024*2);
  h16*   g2pf= (h16*)  alloc((size_t)BT*128*256*2);
  h16*   g1pf= (h16*)  alloc((size_t)BT*512*128*2);
  h16*   g0p = (h16*)  alloc((size_t)BT*Nt*128*2);
  int*   fpsb= (int*)  alloc((size_t)BT*512*4);
  int*   bqb = (int*)  alloc((size_t)BT*512*64*4);
  int*   kni = (int*)  alloc((size_t)BT*Nt*3*4);
  float* knw = (float*)alloc((size_t)BT*Nt*3*4);
  h16* slabA = (h16*)alloc((size_t)131072*288*2);   // 75.5 MB
  h16* slabB = (h16*)alloc((size_t)131072*128*2);   // 33.5 MB

  Ctx C{stream, slabA, slabB, fpsb, bqb, kni, knw};

  // l0 inputs
  { size_t t = (size_t)BT*Nt*3;  k_xyz_cf <<<cdivz(t,256),256,0,stream>>>(pc, xyz0, Nt, t); }
  { size_t t = (size_t)BT*Nt*32; k_feat_cf<<<cdivz(t,256),256,0,stream>>>(pc, feat0, Nt, 3, 32, t); }
  { size_t R = (size_t)BT*Nt;
    k_cat_xyz  <<<cdivz(R*3,256),256,0,stream>>>(xyz0, cat6, 32, R*3);
    k_copy_cols<<<cdivz(R*3,256),256,0,stream>>>(feat0, 32, 3, cat6, 32, 3, R*3); }

  // Pipeline 1
  sa_stage(C, xyz0, Nt, feat0, 32, 3, 512, 64, 0.2f, &Ls[sa01], 3, l1x, l1p);
  sa_stage(C, l1x, 512, l1p, 128, 128, 128, 64, 0.4f, &Ls[sa02], 3, l2x, l2p);
  sa_groupall(C, l2x, 128, l2p, 256, 256, &Ls[sa03], 3, l3p);
  fp_stage(C, l2x, 128, nullptr, 1, l2p, 256, 256, l3p, 1024, 1024, &Ls[fp03], 2, l2pf);
  fp_stage(C, l1x, 512, l2x, 128, l1p, 128, 128, l2pf, 256, 256, &Ls[fp02], 2, l1pf);
  fp_stage(C, xyz0, Nt, l1x, 512, cat6, 32, 6, l1pf, 128, 128, &Ls[fp01], 3, l0p0);

  // Pipeline 2
  sa_stage(C, xyz0, Nt, feat0, 32, 3, 512, 64, 0.3f, &Ls[sa04], 3, m1x, m1p);
  sa_stage(C, m1x, 512, m1p, 128, 128, 128, 64, 0.5f, &Ls[sa05], 3, m2x, m2p);
  sa_groupall(C, m2x, 128, m2p, 256, 256, &Ls[sa06], 3, m3p);
  fp_stage(C, m2x, 128, nullptr, 1, m2p, 256, 256, m3p, 1024, 1024, &Ls[fp04], 2, m2pf);
  fp_stage(C, m1x, 512, m2x, 128, m1p, 128, 128, m2pf, 256, 256, &Ls[fp05], 2, m1pf);
  fp_stage(C, xyz0, Nt, m1x, 512, cat6, 32, 6, m1pf, 128, 128, &Ls[fp06], 3, l0p1);

  // net = concat(conv00(l0p0), conv01(l0p1))  -- write both halves in place
  run_gemm(stream, l0p0, 128, Ls[conv00], net,        256, BT*Nt, 1.f, false, false);
  run_gemm(stream, l0p1, 128, Ls[conv01], net + 128,  256, BT*Nt, 1.f, false, false);

  { size_t R = (size_t)BT*Nt;
    k_cat_xyz  <<<cdivz(R*3,256),256,0,stream>>>(xyz0, cat259, 288, R*3);
    k_copy_cols<<<cdivz(R*256,256),256,0,stream>>>(net, 256, 256, cat259, 288, 3, R*256); }

  // Heatmap branches
  for (int hb = 0; hb < 4; ++hb){
    sa_stage(C, xyz0, Nt, net, 256, 256, 512, 64, 0.2f, &Ls[bsa1[hb]], 3, g1x, g1p);
    sa_stage(C, g1x, 512, g1p, 128, 128, 128, 64, 0.4f, &Ls[bsa2[hb]], 3, g2x, g2p);
    sa_groupall(C, g2x, 128, g2p, 256, 256, &Ls[bsa3[hb]], 3, g3p);
    fp_stage(C, g2x, 128, nullptr, 1, g2p, 256, 256, g3p, 1024, 1024, &Ls[bfp3[hb]], 2, g2pf);
    fp_stage(C, g1x, 512, g2x, 128, g1p, 128, 128, l2pf /*l2_points0*/, 256, 256, &Ls[bfp2[hb]], 2, g1pf);
    fp_stage(C, xyz0, Nt, g1x, 512, cat259, 288, 259, g1pf, 128, 128, &Ls[bfp1[hb]], 3, g0p);

    int M = BT * Nt; size_t R = (size_t)M;
    run_gemm(stream, g0p, 128, Ls[bc0[hb]], slabA, 512, M, 1.f, false, false);      // h0 (512)
    k_cat_xyz  <<<cdivz(R*3,256),256,0,stream>>>(xyz0, slabB, 544, R*3);
    k_copy_cols<<<cdivz(R*512,256),256,0,stream>>>(slabA, 512, 512, slabB, 544, 3, R*512);
    run_gemm(stream, slabB, 544, Ls[bc1[hb]], slabA, 128, M, 1.f, false, false);    // h1 (128)
    k_cat_xyz  <<<cdivz(R*3,256),256,0,stream>>>(xyz0, slabB, 160, R*3);
    k_copy_cols<<<cdivz(R*128,256),256,0,stream>>>(slabA, 128, 128, slabB, 160, 3, R*128);
    run_gemm(stream, slabB, 160, Ls[bc2[hb]], slabA, 32, M, 1.f, false, false);     // h2 (32)
    k_cat_xyz  <<<cdivz(R*3,256),256,0,stream>>>(xyz0, slabB, 64, R*3);
    k_copy_cols<<<cdivz(R*32,256),256,0,stream>>>(slabA, 32, 32, slabB, 64, 3, R*32);
    run_gemm(stream, slabB, 64, Ls[bc3[hb]], slabA, 32, M, 1.f, false, false);      // h3 (1, ld 32)
    run_gemm(stream, slabA, 32, Ls[bc4[hb]],
             (float*)d_out + (size_t)hb * M, 1, M, 1.f, false, true);               // heatmap (f32)
  }
}
